// SpatialMemoryCore_63926293233725
// MI455X (gfx1250) — matrix operand order repaired and checked by reading, not compiled, not run on hardware
//
#include <hip/hip_runtime.h>

// ---------------------------------------------------------------------------
// SpatialMemoryCore for MI455X (gfx1250, wave32, WMMA 16x16x32 f16)
//  - convs fused per-image in ~301KB LDS (halo-padded x tile, y1 resident)
//  - async global->LDS staging (ASYNCcnt) for the conv input tile
//  - MLPs / final linear as generic WMMA GEMM (fp16 in, fp32 acc)
//  - scatter path with LDS-staged gate MLP + global fp32 atomics
// ---------------------------------------------------------------------------

typedef __attribute__((ext_vector_type(16))) _Float16 v16h;
typedef __attribute__((ext_vector_type(8)))  _Float16 v8h;
typedef __attribute__((ext_vector_type(8)))  float    v8f;

#define NB    2048
#define NF    512
#define NM    16
#define NHW   1024      // 32*32
#define XD    576       // F + PD
#define NK    121       // 11*11
#define SIG2X 26.888888888888889f   // 2*(11/3)^2
#define LN1E4_OVER16 0.5756462732485115f

__device__ __forceinline__ v16h cat16(v8h lo, v8h hi) {
  return __builtin_shufflevector(lo, hi, 0,1,2,3,4,5,6,7,8,9,10,11,12,13,14,15);
}
__device__ __forceinline__ v8f wmma_f16(v16h a, v16h b, v8f c) {
  return __builtin_amdgcn_wmma_f32_16x16x32_f16(false, a, false, b, (short)0, c, false, false);
}
// low 32 bits of a generic pointer into LDS == wave-relative LDS byte address
__device__ __forceinline__ unsigned lds_addr32(const void* p) {
  return (unsigned)(uintptr_t)p;
}

// ---------------------------------------------------------------------------
// K0: positional encoding + concat [features | pe]  -> xcat (B x 576) f16
// ---------------------------------------------------------------------------
__global__ __launch_bounds__(128) void k_posenc(const float* __restrict__ feat,
                                                const float* __restrict__ gaze,
                                                _Float16* __restrict__ xcat) {
  int b = blockIdx.x, t = threadIdx.x;
  _Float16* row = xcat + (size_t)b * XD;
  const float* frow = feat + (size_t)b * NF;
  for (int i = t; i < NF; i += 128) row[i] = (_Float16)frow[i];
  if (t < 64) {
    int axis = t >> 5;          // 0 = x, 1 = y
    int q    = t & 31;          // position within axis block
    int i    = q >> 1;          // frequency index 0..15
    float p  = gaze[b * 2 + axis];
    float dv = __expf(-(float)i * LN1E4_OVER16);
    float v  = p * dv;
    row[NF + t] = (_Float16)((q & 1) ? __cosf(v) : __sinf(v));
  }
}

// ---------------------------------------------------------------------------
// Weight prepack: W (K x N, row-major fp32) -> Bt (N x K, fp16)
// ---------------------------------------------------------------------------
__global__ void k_prep_bt(const float* __restrict__ W, _Float16* __restrict__ Bt,
                          int K, int N) {
  int total = K * N;
  for (int i = blockIdx.x * blockDim.x + threadIdx.x; i < total;
       i += gridDim.x * blockDim.x) {
    int n = i / K, k = i - n * K;
    Bt[(size_t)n * K + k] = (_Float16)W[(size_t)k * N + n];
  }
}

// conv1 weights (128,16,3,3) -> B1[oc][k], k = tap*16+ic, padded to K=160
__global__ void k_prep_c1(const float* __restrict__ ck1, _Float16* __restrict__ B1) {
  int i = blockIdx.x * blockDim.x + threadIdx.x;          // 128*160
  if (i >= 128 * 160) return;
  int oc = i / 160, k = i - oc * 160;
  int tap = k >> 4, ic = k & 15;
  float v = 0.f;
  if (tap < 9) v = ck1[((oc * 16 + ic) * 3 + tap / 3) * 3 + tap % 3];
  B1[i] = (_Float16)v;
}

// conv2 weights (64,128,3,3) -> B2[oc][k], k = tap*128+ic, K=1152
__global__ void k_prep_c2(const float* __restrict__ ck2, _Float16* __restrict__ B2) {
  int i = blockIdx.x * blockDim.x + threadIdx.x;          // 64*1152
  if (i >= 64 * 1152) return;
  int oc = i / 1152, k = i - oc * 1152;
  int tap = k >> 7, ic = k & 127;
  B2[i] = (_Float16)ck2[((oc * 128 + ic) * 3 + tap / 3) * 3 + tap % 3];
}

// ---------------------------------------------------------------------------
// Generic WMMA GEMM: C(2048 x N) = act(A(2048 x K) @ Bt^T + bias)
// A row-major f16, Bt is N x K f16 (weights transposed). One wave per tile.
// ---------------------------------------------------------------------------
template<bool RELU, bool F16OUT>
__global__ __launch_bounds__(32) void k_gemm(const _Float16* __restrict__ A,
                                             const _Float16* __restrict__ Bt,
                                             const float* __restrict__ bias,
                                             _Float16* __restrict__ Ch,
                                             float* __restrict__ Cf,
                                             int N, int K) {
  int mt = blockIdx.x, nt = blockIdx.y;
  int lane = threadIdx.x;
  int hf = lane >> 4, l = lane & 15;
  int m = mt * 16 + l;
  int n = nt * 16 + l;
  v8f acc = {};
  for (int kb = 0; kb < K; kb += 32) {
    int ka = kb + hf * 16;                   // A: K {0..15}/{16..31}
    v8h alo = *(const v8h*)(A + (size_t)m * K + ka);
    v8h ahi = *(const v8h*)(A + (size_t)m * K + ka + 8);
    v16h bfrag = *(const v16h*)(Bt + (size_t)n * K + kb + hf * 16); // B: 16 consec K
    acc = wmma_f16(cat16(alo, ahi), bfrag, acc);
  }
  float bn = bias[n];
#pragma unroll
  for (int r = 0; r < 8; ++r) {
    int mr = mt * 16 + r + hf * 8;
    float v = acc[r] + bn;
    if (RELU) v = fmaxf(v, 0.f);
    if (F16OUT) Ch[(size_t)mr * N + n] = (_Float16)v;
    else        Cf[(size_t)mr * N + n] = v;
  }
}

// ---------------------------------------------------------------------------
// K4: per-(b,k) geometry + gather + strength-gate MLP + atomic scatter
// one workgroup (128 threads) per batch element; k = threadIdx.x < 121
// ---------------------------------------------------------------------------
__global__ __launch_bounds__(128) void k_scatter(const float* __restrict__ cell,
                                                 const float* __restrict__ gaze,
                                                 const float* __restrict__ wvals,
                                                 const float* __restrict__ ws1,
                                                 const float* __restrict__ bs1,
                                                 const float* __restrict__ ws2,
                                                 const float* __restrict__ bs2,
                                                 float* __restrict__ den,
                                                 float* __restrict__ outUpd) {
  __shared__ float sW1[32 * 64];
  __shared__ float sB1[64];
  __shared__ float sW2[64];
  __shared__ float red[128];
  __shared__ float sB2;
  int b = blockIdx.x, t = threadIdx.x;
  for (int i = t; i < 32 * 64; i += 128) sW1[i] = ws1[i];
  if (t < 64) { sB1[t] = bs1[t]; sW2[t] = ws2[t]; }
  if (t == 0) sB2 = bs2[0];

  float gx = fminf(fmaxf(gaze[b * 2 + 0] * 31.f, 0.f), 31.f);
  float gy = fminf(fmaxf(gaze[b * 2 + 1] * 31.f, 0.f), 31.f);
  int x0 = (int)floorf(gx), y0 = (int)floorf(gy);

  float gauss = 0.f; int xi = 0, yi = 0;
  if (t < NK) {
    int ox = t % 11 - 5, oy = t / 11 - 5;
    xi = min(max(x0 + ox, 0), 31);
    yi = min(max(y0 + oy, 0), 31);
    float dx = (float)xi - gx, dy = (float)yi - gy;
    gauss = __expf(-(dx * dx + dy * dy) / SIG2X);
  }
  red[t] = gauss;
  __syncthreads();
  for (int s = 64; s > 0; s >>= 1) {
    if (t < s) red[t] += red[t + s];
    __syncthreads();
  }
  float nw = gauss / fmaxf(red[0], 1e-8f);

  if (t < NK) {
    int idx = yi * 32 + xi;
    float prev[NM], wv[NM];
#pragma unroll
    for (int m = 0; m < NM; ++m) prev[m] = cell[(size_t)b * NM * NHW + m * NHW + idx];
#pragma unroll
    for (int m = 0; m < NM; ++m) wv[m] = wvals[(size_t)b * (NK * NM) + t * NM + m];
    // strength head: relu([prev,wv] @ ws1 + bs1) @ ws2 + bs2 -> sigmoid
    float z = sB2;
    for (int h = 0; h < 64; ++h) {
      float s = sB1[h];
#pragma unroll
      for (int i = 0; i < NM; ++i) s += prev[i] * sW1[i * 64 + h];
#pragma unroll
      for (int i = 0; i < NM; ++i) s += wv[i] * sW1[(NM + i) * 64 + h];
      z += fmaxf(s, 0.f) * sW2[h];
    }
    float gstr = 1.f / (1.f + __expf(-z));
    float w = nw * gstr;
    atomicAdd(&den[(size_t)b * NHW + idx], w);
#pragma unroll
    for (int m = 0; m < NM; ++m) {
      float nv = (1.f - w) * prev[m] + w * wv[m];
      atomicAdd(&outUpd[(size_t)b * NM * NHW + m * NHW + idx], w * nv);
    }
  }
}

// K5: updated = keep*cell + num (num accumulated in outUpd); also emit the
// fp16 [b][pix][ic] copy consumed by the async-LDS conv staging.
__global__ __launch_bounds__(256) void k_finalize(const float* __restrict__ cell,
                                                  const float* __restrict__ den,
                                                  float* __restrict__ outUpd,
                                                  _Float16* __restrict__ updH) {
  int p = blockIdx.x * blockDim.x + threadIdx.x;
  if (p >= NB * NHW) return;
  int b = p >> 10, pix = p & (NHW - 1);
  float keep = 1.f - fminf(den[p], 1.f);
  _Float16* hrow = updH + ((size_t)b * NHW + pix) * NM;
#pragma unroll
  for (int m = 0; m < NM; ++m) {
    size_t o = (size_t)b * NM * NHW + m * NHW + pix;
    float v = keep * cell[o] + outUpd[o];
    outUpd[o] = v;
    hrow[m] = (_Float16)v;
  }
}

// ---------------------------------------------------------------------------
// Fused conv1(16->128) + conv2(128->64) + adaptive pool 3x3, one WG per image
// LDS: x halo-padded 34x34x16 f16 (36.1KB, zero border -> branchless conv1)
//      + y1 1024x128 f16 (256KB) + pool bins.  Input tile staged with
//      GLOBAL_LOAD_ASYNC_TO_LDS_B128 (ASYNCcnt) from the fp16 [pix][ic] copy.
// ---------------------------------------------------------------------------
#define XP 34  // padded tile edge

__device__ __forceinline__ v8h tap_ld(const _Float16* xs, int py, int px,
                                      int tap, int icb) {
  v8h z = {};
  if (tap >= 9) return z;                       // folds under full unroll
  int sy = py + tap / 3, sx = px + tap % 3;     // halo of 1 absorbs the -1
  return *(const v8h*)(xs + (sy * XP + sx) * 16 + icb);
}

__global__ __launch_bounds__(256) void k_conv_fused(const _Float16* __restrict__ updH,
                                                    const _Float16* __restrict__ B1,
                                                    const _Float16* __restrict__ B2,
                                                    const float* __restrict__ cb1,
                                                    const float* __restrict__ cb2,
                                                    _Float16* __restrict__ pooled) {
  __shared__ _Float16 xs[XP * XP * 16];    //  36.1 KB (halo-padded)
  __shared__ _Float16 y1[NHW * 128];       // 256 KB
  __shared__ float    pool[576];           //  2.25 KB

  int b = blockIdx.x, t = threadIdx.x;
  int lane = t & 31, wv = t >> 5;          // 8 waves
  int hf = lane >> 4, l = lane & 15;

  // prefetch conv2 weight pack (L2-hot across all 2048 workgroups)
  for (int i = t * 256; i < 64 * 1152 * 2; i += 256 * 256)
    __builtin_prefetch((const char*)B2 + i, 0, 3);

  // async-stage interior of x tile: 1024 px * 32B = 2048 x b128 transfers
  const _Float16* src = updH + (size_t)b * (NHW * NM);
  for (int i = t; i < 2048; i += 256) {
    int p = i >> 1, part = i & 1;
    unsigned dst = lds_addr32(
        &xs[(((p >> 5) + 1) * XP + (p & 31) + 1) * 16 + part * 8]);
    const void* g = src + p * 16 + part * 8;
    asm volatile("global_load_async_to_lds_b128 %0, %1, off"
                 :: "v"(dst), "v"(g) : "memory");
  }
  // zero the halo border
  for (int i = t; i < XP * XP; i += 256) {
    int r = i / XP, c = i - r * XP;
    if (r == 0 || r == XP - 1 || c == 0 || c == XP - 1) {
      v8h z = {};
      *(v8h*)&xs[i * 16] = z;
      *(v8h*)&xs[i * 16 + 8] = z;
    }
  }
  if (t < 576) pool[t] = 0.f;

  // conv1 B fragments: each wave owns one 16-wide output-channel tile
  int n1 = wv * 16 + l;
  v16h b1f[5];
#pragma unroll
  for (int kb = 0; kb < 5; ++kb)
    b1f[kb] = *(const v16h*)(B1 + n1 * 160 + kb * 32 + hf * 16);
  float bias1 = cb1[n1];

  asm volatile("s_wait_asynccnt 0" ::: "memory");
  __syncthreads();

  // conv1: 64 pixel tiles, branchless halo-padded im2col from LDS
  for (int mt = 0; mt < 64; ++mt) {
    int p = mt * 16 + l, py = p >> 5, px = p & 31;
    v8f acc = {};
#pragma unroll
    for (int kb = 0; kb < 5; ++kb) {
      v8h alo = tap_ld(xs, py, px, kb * 2 + hf, 0);
      v8h ahi = tap_ld(xs, py, px, kb * 2 + hf, 8);
      acc = wmma_f16(cat16(alo, ahi), b1f[kb], acc);
    }
#pragma unroll
    for (int r = 0; r < 8; ++r) {
      int pr = mt * 16 + r + hf * 8;
      y1[pr * 128 + n1] = (_Float16)fmaxf(acc[r] + bias1, 0.f);
    }
  }
  __syncthreads();

  // conv2: 4 out-channel tiles x 64 pixel tiles, 2 waves share each n-tile;
  // bounds check hoisted to one predicate per tap (4 WMMAs per tap)
  int n2 = (wv & 3) * 16 + l;
  float bias2 = cb2[n2];
  for (int mt = (wv >> 2); mt < 64; mt += 2) {
    int p = mt * 16 + l, py = p >> 5, px = p & 31;
    v8f acc = {};
#pragma unroll
    for (int tap = 0; tap < 9; ++tap) {
      int sy = py + tap / 3 - 1, sx = px + tap % 3 - 1;
      bool inb = (unsigned)sy < 32u && (unsigned)sx < 32u;
      const _Float16* base = y1 + (sy * 32 + sx) * 128;
#pragma unroll
      for (int j = 0; j < 4; ++j) {
        int icb = j * 32 + hf * 16;
        v8h alo = {}, ahi = {};
        if (inb) {
          alo = *(const v8h*)(base + icb);
          ahi = *(const v8h*)(base + icb + 8);
        }
        v16h bf = *(const v16h*)(B2 + n2 * 1152 + (tap * 4 + j) * 32 + hf * 16);
        acc = wmma_f16(cat16(alo, ahi), bf, acc);
      }
    }
#pragma unroll
    for (int r = 0; r < 8; ++r) {
      float v = fmaxf(acc[r] + bias2, 0.f);
      int pr = mt * 16 + r + hf * 8;
      int py2 = pr >> 5, px2 = pr & 31;
      // overlapping adaptive-pool bins: [0,11) [10,22) [21,32)
      int rb[2], cb[2], nr = 1, nc = 1;
      rb[0] = (py2 <= 10) ? 0 : ((py2 <= 21) ? 1 : 2);
      if (py2 == 10) { rb[1] = 1; nr = 2; } else if (py2 == 21) { rb[1] = 2; nr = 2; }
      cb[0] = (px2 <= 10) ? 0 : ((px2 <= 21) ? 1 : 2);
      if (px2 == 10) { cb[1] = 1; nc = 2; } else if (px2 == 21) { cb[1] = 2; nc = 2; }
      for (int a = 0; a < nr; ++a)
        for (int c = 0; c < nc; ++c)
          atomicAdd(&pool[n2 * 9 + rb[a] * 3 + cb[c]], v);
    }
  }
  __syncthreads();
  if (t < 576) {
    const int rows[3] = {11, 12, 11};
    int bin = t % 9;
    pooled[(size_t)b * 576 + t] =
        (_Float16)(pool[t] / (float)(rows[bin / 3] * rows[bin % 3]));
  }
}

// ---------------------------------------------------------------------------
extern "C" void kernel_launch(void* const* d_in, const int* in_sizes, int n_in,
                              void* d_out, int out_size, void* d_ws, size_t ws_size,
                              hipStream_t stream) {
  const float* features = (const float*)d_in[0];
  const float* cell     = (const float*)d_in[1];
  const float* gaze     = (const float*)d_in[2];
  const float* w1  = (const float*)d_in[3];  const float* b1  = (const float*)d_in[4];
  const float* w2  = (const float*)d_in[5];  const float* b2  = (const float*)d_in[6];
  const float* wvW = (const float*)d_in[7];  const float* bv  = (const float*)d_in[8];
  const float* ws1 = (const float*)d_in[9];  const float* bs1 = (const float*)d_in[10];
  const float* ws2 = (const float*)d_in[11]; const float* bs2 = (const float*)d_in[12];
  const float* ck1 = (const float*)d_in[13]; const float* cb1 = (const float*)d_in[14];
  const float* ck2 = (const float*)d_in[15]; const float* cb2 = (const float*)d_in[16];
  const float* wo  = (const float*)d_in[17]; const float* bo  = (const float*)d_in[18];

  float* out    = (float*)d_out;                   // (B, 576)
  float* outUpd = (float*)d_out + (size_t)NB * XD; // (B, 16, 32, 32)

  // ---- workspace carve-out (256B aligned) ----
  char* wsb = (char*)d_ws;
  size_t off = 0;
  auto carve = [&](size_t bytes) {
    char* p = wsb + off;
    off = (off + bytes + 255) & ~(size_t)255;
    return p;
  };
  _Float16* xcat  = (_Float16*)carve((size_t)NB * XD * 2);
  _Float16* g1    = (_Float16*)carve((size_t)NB * 256 * 2);
  _Float16* g2    = (_Float16*)carve((size_t)NB * 128 * 2);
  float*    wvout = (float*)   carve((size_t)NB * (NK * NM) * 4);
  float*    den   = (float*)   carve((size_t)NB * NHW * 4);
  _Float16* pooled= (_Float16*)carve((size_t)NB * XD * 2);
  _Float16* updH  = (_Float16*)carve((size_t)NB * NHW * NM * 2);
  _Float16* w1T   = (_Float16*)carve((size_t)256 * XD * 2);
  _Float16* w2T   = (_Float16*)carve((size_t)128 * 256 * 2);
  _Float16* wvT   = (_Float16*)carve((size_t)(NK * NM) * 128 * 2);
  _Float16* woT   = (_Float16*)carve((size_t)XD * XD * 2);
  _Float16* B1c   = (_Float16*)carve((size_t)128 * 160 * 2);
  _Float16* B2c   = (_Float16*)carve((size_t)64 * 1152 * 2);
  (void)ws_size; (void)in_sizes; (void)n_in; (void)out_size;

  // zero the accumulator regions (graph-capture-safe)
  hipMemsetAsync(outUpd, 0, (size_t)NB * NM * NHW * 4, stream);
  hipMemsetAsync(den, 0, (size_t)NB * NHW * 4, stream);

  // weight prepacks
  k_prep_bt<<<512, 256, 0, stream>>>(w1, w1T, XD, 256);
  k_prep_bt<<<128, 256, 0, stream>>>(w2, w2T, 256, 128);
  k_prep_bt<<<512, 256, 0, stream>>>(wvW, wvT, 128, NK * NM);
  k_prep_bt<<<512, 256, 0, stream>>>(wo, woT, XD, XD);
  k_prep_c1<<<(128 * 160 + 255) / 256, 256, 0, stream>>>(ck1, B1c);
  k_prep_c2<<<(64 * 1152 + 255) / 256, 256, 0, stream>>>(ck2, B2c);

  // input gate + write values
  k_posenc<<<NB, 128, 0, stream>>>(features, gaze, xcat);
  k_gemm<true,  true ><<<dim3(NB / 16, 256 / 16), 32, 0, stream>>>(xcat, w1T, b1, g1, nullptr, 256, XD);
  k_gemm<true,  true ><<<dim3(NB / 16, 128 / 16), 32, 0, stream>>>(g1, w2T, b2, g2, nullptr, 128, 256);
  k_gemm<false, false><<<dim3(NB / 16, (NK * NM) / 16), 32, 0, stream>>>(g2, wvT, bv, nullptr, wvout, NK * NM, 128);

  // gather / gate / scatter, then finalize updated cell state (+f16 copy)
  k_scatter<<<NB, 128, 0, stream>>>(cell, gaze, wvout, ws1, bs1, ws2, bs2, den, outUpd);
  k_finalize<<<(NB * NHW + 255) / 256, 256, 0, stream>>>(cell, den, outUpd, updH);

  // fused output-gate convs + adaptive pool (LDS-resident per image)
  k_conv_fused<<<NB, 256, 0, stream>>>(updH, B1c, B2c, cb1, cb2, pooled);

  // final linear: out = pooled @ wo + bo
  k_gemm<false, false><<<dim3(NB / 16, XD / 16), 32, 0, stream>>>(pooled, woT, bo, nullptr, out, XD, XD);
}
